// BatchDistanceAwareCrossAttention_44702019616956
// MI455X (gfx1250) — matrix operand order repaired and checked by reading, not compiled, not run on hardware
//
#include <hip/hip_runtime.h>
#include <hip/hip_bf16.h>
#include <stddef.h>

// ---------------------------------------------------------------------------
// Problem constants
// ---------------------------------------------------------------------------
#define B_SZ   32
#define N_CTX  4096
#define D_DIM  1024
#define HEADS  16
#define HEAD_D 64
#define SPLITS 16            // N split into 16 chunks of 256
#define CHUNK  256           // context rows per flash block
#define KT     32            // WMMA K tile (bf16 -> K=32)
#define LAMBDA (-0.1f)
#define LN_EPS 1e-5f
#define SCALE  0.125f        // 1/sqrt(64)

typedef __attribute__((ext_vector_type(16))) __bf16        v16bf;
typedef __attribute__((ext_vector_type(8)))  float         v8f;
typedef __attribute__((ext_vector_type(4)))  unsigned int  u32x4;
typedef __attribute__((ext_vector_type(8)))  int           i32x8;
typedef __attribute__((ext_vector_type(4)))  int           i32x4;

// f32 -> bf16: native convert (hardware bf16 cvt on gfx1250)
static __device__ __forceinline__ __bf16 f2bf(float f) { return (__bf16)f; }
static __device__ __forceinline__ unsigned short f2bfu(float f) {
    union { __bf16 b; unsigned short u; } y; y.b = (__bf16)f; return y.u;
}
static __device__ __forceinline__ __bf16 u2bf(unsigned short s) {
    union { unsigned short u; __bf16 b; } y; y.u = s; return y.b;
}

// K index within a 32-wide tile for the 16-bit A-matrix layout (ISA 7.12.2):
// VGPR r holds K = {2r, 2r+1} (r<4) or {2r+8, 2r+9} (r>=4); lanes 16..31 add +8.
static __device__ __forceinline__ int a_k_index(int e, int lane) {
    int r = e >> 1;
    int k = ((r < 4) ? (2 * r) : (2 * r + 8)) + (e & 1);
    return k + ((lane & 16) ? 8 : 0);
}

// ---------------------------------------------------------------------------
// Tensor Data Mover: 2D f32 tile (tile_w x tile_h, row stride in elements)
// global -> LDS, with LDS padding inserted by the DMA (pad 2 DWORDs every
// `pad_interval_code` DWORDs).  D# built per CDNA5 ISA sec 8.3-8.6.
// clang-23 toolchain: 6-arg builtin (g0 u32x4, g1 i32x8, g2 i32x4, g3 i32x4,
// extra i32x8, cpol).  Groups 2/3 + extra are zero (2D tensor, no gather).
// Issue from ONE wave only; EXEC is ignored by tensor ops.
// ---------------------------------------------------------------------------
static __device__ __forceinline__ void tdm_load_2d(unsigned lds_addr,
                                                   const float* gaddr,
                                                   unsigned tile_w,
                                                   unsigned tile_h,
                                                   unsigned row_stride,
                                                   unsigned tdim0,
                                                   unsigned tdim1,
                                                   unsigned pad_interval_code) {
    unsigned long long ga = (unsigned long long)(uintptr_t)gaddr;
    u32x4 g0;
    g0[0] = 1u;                                            // count=1, user D#
    g0[1] = lds_addr;                                      // LDS byte address
    g0[2] = (unsigned)(ga & 0xFFFFFFFFu);                  // global_addr lo
    g0[3] = (unsigned)((ga >> 32) & 0x01FFFFFFu) | (2u << 30);  // addr hi | type=2
    i32x8 g1;
    // data_size=4B(code 2) | pad_enable | pad_interval | pad_amount(code1 = 2 DW)
    g1[0] = (int)((2u << 16) | (1u << 20) | (pad_interval_code << 22) | (1u << 25));
    g1[1] = (int)((tdim0 & 0xFFFFu) << 16);                // tensor_dim0 lo16
    g1[2] = (int)(((tdim0 >> 16) & 0xFFFFu) | ((tdim1 & 0xFFFFu) << 16));
    g1[3] = (int)(((tdim1 >> 16) & 0xFFFFu) | ((tile_w & 0xFFFFu) << 16));
    g1[4] = (int)(tile_h & 0xFFFFu);                       // tile_dim1, tile_dim2=0
    g1[5] = (int)row_stride;                               // tensor_dim0_stride lo32
    g1[6] = 0;                                             // stride0 hi16 | stride1 lo16
    g1[7] = 0;                                             // stride1 hi32
    i32x4 z4 = { 0, 0, 0, 0 };
    i32x8 z8 = { 0, 0, 0, 0, 0, 0, 0, 0 };
    __builtin_amdgcn_tensor_load_to_lds(g0, g1, z4, z4, z8, 0);
}

// ---------------------------------------------------------------------------
// K0: q projection  qp[b,:] = query[b,:] @ Wq + bq            (tiny GEMV)
// ---------------------------------------------------------------------------
__global__ __launch_bounds__(256) void k_qproj(const float* __restrict__ query,
                                               const float* __restrict__ Wq,
                                               const float* __restrict__ bq,
                                               float* __restrict__ qp) {
    const int b = blockIdx.x;
    const int dout = threadIdx.x * 4;
    float4 acc = { bq[dout], bq[dout + 1], bq[dout + 2], bq[dout + 3] };
    const float* qrow = query + (size_t)b * D_DIM;
    for (int din = 0; din < D_DIM; ++din) {
        float qv = qrow[din];
        const float4 w = *(const float4*)(Wq + (size_t)din * D_DIM + dout);
        acc.x += qv * w.x; acc.y += qv * w.y; acc.z += qv * w.z; acc.w += qv * w.w;
    }
    *(float4*)(qp + (size_t)b * D_DIM + dout) = acc;
}

// ---------------------------------------------------------------------------
// K1: u matrix in WMMA-A bf16 fragment layout.
//  u[b,h,din] = SCALE * sum_j Wk[din, h*64+j] * qp[b, h*64+j]
// ---------------------------------------------------------------------------
__global__ __launch_bounds__(256) void k_uA(const float* __restrict__ Wk,
                                            const float* __restrict__ qp,
                                            unsigned short* __restrict__ uA) {
    const int idx = blockIdx.x * 256 + threadIdx.x;    // [0, 32*32*32*16)
    const int e = idx & 15;
    const int l = (idx >> 4) & 31;
    const int t = (idx >> 9) & 31;
    const int b = idx >> 14;
    const int m = l & 15;
    const int K = t * 32 + a_k_index(e, l);
    const float* qh = qp + (size_t)b * D_DIM + m * HEAD_D;
    const float* wr = Wk + (size_t)K * D_DIM + m * HEAD_D;
    float acc = 0.f;
    for (int j = 0; j < HEAD_D; ++j) acc += wr[j] * qh[j];
    uA[idx] = f2bfu(acc * SCALE);
}

// per-(b,h) constant from bk:  kc = SCALE * dot(bk[hslice], qp[b,hslice])
__global__ void k_kconst(const float* __restrict__ bk,
                         const float* __restrict__ qp,
                         float* __restrict__ kconst) {
    const int i = threadIdx.x;           // 512 = 32*16
    if (i >= B_SZ * HEADS) return;
    const int b = i >> 4, h = i & 15;
    float acc = 0.f;
    for (int j = 0; j < HEAD_D; ++j)
        acc += bk[h * HEAD_D + j] * qp[(size_t)b * D_DIM + h * HEAD_D + j];
    kconst[i] = acc * SCALE;
}

// ---------------------------------------------------------------------------
// K2: flash split-softmax attention core.  grid = (SPLITS, B), 256 thr (8 waves)
//  Context tiles are DMAed into LDS by the Tensor Data Mover (f32, padded by
//  the DMA), converted to bf16 at fragment-build time, and fed to WMMA.
// ---------------------------------------------------------------------------
#define TSA 34               // phase-A LDS row stride in f32 ([n][k]): 32 + 2 pad
#define TSB 258              // phase-B LDS row stride in f32 ([k][d]): 256 + 2 pad
#define PST 264              // P row stride (bf16)

__global__ __launch_bounds__(256) void k_flash(const float* __restrict__ ctx,
                                               const float* __restrict__ dist,
                                               const int*   __restrict__ mask,
                                               const unsigned short* __restrict__ uA,
                                               const float* __restrict__ kconst,
                                               float* __restrict__ mpart,
                                               float* __restrict__ lpart,
                                               float* __restrict__ wpart) {
    __shared__ float tileF[CHUNK * TSA];           // 34816 B; phase B uses 32*TSB=8256 f32
    __shared__ unsigned short plds[HEADS * PST];   // softmax probabilities, bf16
    __shared__ float red[256];                     // [head][tile] partials
    __shared__ float smax[HEADS];

    const int b = blockIdx.y, sp = blockIdx.x;
    const int n0 = sp * CHUNK;
    const int tid = threadIdx.x;
    const int lane = tid & 31, wave = tid >> 5;
    const int colL = lane & 15;
    const int khalf = lane >> 4;
    const int moff = (lane & 16) ? 8 : 0;          // C layout: M = r + moff
    const float* ctxb = ctx + (size_t)b * N_CTX * D_DIM;
    const unsigned tile_lds = (unsigned)(uintptr_t)&tileF[0];

    // ---------------- Phase A: scores = u x ctx^T ----------------
    v8f c0 = {}; v8f c1 = {};
    for (int t = 0; t < D_DIM / KT; ++t) {
        if (wave == 0) {
            // TDM: ctx[n0..n0+255, t*32..+32) f32 -> LDS [n][k], stride 34
            // pad_interval code 4 = every 32 DWORDs, pad 2 DWORDs
            tdm_load_2d(tile_lds, ctxb + (size_t)n0 * D_DIM + t * KT,
                        /*w*/KT, /*h*/CHUNK, /*stride*/D_DIM,
                        /*tdim0*/D_DIM, /*tdim1*/N_CTX, /*pad_code*/4);
            __builtin_amdgcn_s_wait_tensorcnt(0);
        } else if (t + 1 < D_DIM / KT) {
            // keep the next k-tile flowing toward L2 (global_prefetch_b8)
            __builtin_prefetch(ctxb + (size_t)(n0 + tid) * D_DIM + (t + 1) * KT, 0, 2);
        }
        __syncthreads();
        // A fragment: 32 contiguous bytes per lane from precomputed uA
        v16bf a;
        {
            const uint4* ap = (const uint4*)(uA + (((size_t)b * 32 + t) * 32 + lane) * 16);
            uint4 w0 = ap[0], w1 = ap[1];
            unsigned wd[8] = { w0.x, w0.y, w0.z, w0.w, w1.x, w1.y, w1.z, w1.w };
            #pragma unroll
            for (int e = 0; e < 16; ++e)
                a[e] = u2bf((unsigned short)(wd[e >> 1] >> (16 * (e & 1))));
        }
        // B fragments (convert f32 -> bf16 here; each element converted once)
        v16bf b0, b1;
        {
            const int nA = (wave * 2 + 0) * 16 + colL;
            const int nB = (wave * 2 + 1) * 16 + colL;
            #pragma unroll
            for (int e = 0; e < 16; ++e) {
                const int k = a_k_index(e, lane);
                b0[e] = f2bf(tileF[nA * TSA + k]);
                b1[e] = f2bf(tileF[nB * TSA + k]);
            }
        }
        c0 = __builtin_amdgcn_wmma_f32_16x16x32_bf16(false, a, false, b0, (short)0, c0, false, false);
        c1 = __builtin_amdgcn_wmma_f32_16x16x32_bf16(false, a, false, b1, (short)0, c1, false, false);
        __syncthreads();
    }

    // bias + mask (C layout: M = r + moff, col = colL)
    const int ng0 = n0 + (wave * 2 + 0) * 16 + colL;
    const int ng1 = n0 + (wave * 2 + 1) * 16 + colL;
    const float db0 = LAMBDA * dist[(size_t)b * N_CTX + ng0];
    const float db1 = LAMBDA * dist[(size_t)b * N_CTX + ng1];
    const bool ok0 = mask[(size_t)b * N_CTX + ng0] != 0;
    const bool ok1 = mask[(size_t)b * N_CTX + ng1] != 0;
    float s0[8], s1[8];
    #pragma unroll
    for (int r = 0; r < 8; ++r) {
        const float kc = kconst[b * HEADS + r + moff];
        s0[r] = ok0 ? (c0[r] + kc + db0) : -1e30f;
        s1[r] = ok1 ? (c1[r] + kc + db1) : -1e30f;
    }
    // row-max: reduce over the 16 lanes of each half-wave
    #pragma unroll
    for (int r = 0; r < 8; ++r) {
        float x = s0[r];
        x = fmaxf(x, __shfl_xor(x, 1, 32)); x = fmaxf(x, __shfl_xor(x, 2, 32));
        x = fmaxf(x, __shfl_xor(x, 4, 32)); x = fmaxf(x, __shfl_xor(x, 8, 32));
        if (colL == 0) red[(r + moff) * 16 + wave * 2 + 0] = x;
        float y = s1[r];
        y = fmaxf(y, __shfl_xor(y, 1, 32)); y = fmaxf(y, __shfl_xor(y, 2, 32));
        y = fmaxf(y, __shfl_xor(y, 4, 32)); y = fmaxf(y, __shfl_xor(y, 8, 32));
        if (colL == 0) red[(r + moff) * 16 + wave * 2 + 1] = y;
    }
    __syncthreads();
    if (tid < HEADS) {
        float mx = red[tid * 16];
        for (int i = 1; i < 16; ++i) mx = fmaxf(mx, red[tid * 16 + i]);
        smax[tid] = mx;
    }
    __syncthreads();
    // p = exp(s - m), stash bf16 P for phase B, reduce row-sums
    float p0[8], p1[8];
    #pragma unroll
    for (int r = 0; r < 8; ++r) {
        const int m = r + moff;
        p0[r] = __expf(s0[r] - smax[m]);
        p1[r] = __expf(s1[r] - smax[m]);
        plds[m * PST + (wave * 2 + 0) * 16 + colL] = f2bfu(p0[r]);
        plds[m * PST + (wave * 2 + 1) * 16 + colL] = f2bfu(p1[r]);
    }
    #pragma unroll
    for (int r = 0; r < 8; ++r) {
        float x = p0[r];
        x += __shfl_xor(x, 1, 32); x += __shfl_xor(x, 2, 32);
        x += __shfl_xor(x, 4, 32); x += __shfl_xor(x, 8, 32);
        if (colL == 0) red[(r + moff) * 16 + wave * 2 + 0] = x;
        float y = p1[r];
        y += __shfl_xor(y, 1, 32); y += __shfl_xor(y, 2, 32);
        y += __shfl_xor(y, 4, 32); y += __shfl_xor(y, 8, 32);
        if (colL == 0) red[(r + moff) * 16 + wave * 2 + 1] = y;
    }
    __syncthreads();
    if (tid < HEADS) {
        float sm = 0.f;
        for (int i = 0; i < 16; ++i) sm += red[tid * 16 + i];
        mpart[(b * SPLITS + sp) * HEADS + tid] = smax[tid];
        lpart[(b * SPLITS + sp) * HEADS + tid] = sm;
    }
    __syncthreads();

    // ---------------- Phase B: wpart = P x ctx_chunk ----------------
    for (int cg = 0; cg < 4; ++cg) {                   // 4 x 256 output cols
        const int d0c = cg * 256;
        v8f a0 = {}; v8f a1 = {};
        for (int kk = 0; kk < CHUNK / KT; ++kk) {      // K over context rows
            if (wave == 0) {
                // TDM: ctx[n0+kk*32 .. +32, d0c..+256) -> LDS [k][d], stride 258
                // pad_interval code 7 = every 256 DWORDs, pad 2 DWORDs
                tdm_load_2d(tile_lds, ctxb + (size_t)(n0 + kk * KT) * D_DIM + d0c,
                            /*w*/256, /*h*/KT, /*stride*/D_DIM,
                            /*tdim0*/D_DIM, /*tdim1*/N_CTX, /*pad_code*/7);
                __builtin_amdgcn_s_wait_tensorcnt(0);
            }
            __syncthreads();
            v16bf a;                                   // A = P fragment (bf16 in LDS)
            #pragma unroll
            for (int e = 0; e < 16; ++e) {
                const int k = kk * KT + a_k_index(e, lane);
                a[e] = u2bf(plds[(lane & 15) * PST + k]);
            }
            v16bf b0, b1;                              // B = ctx fragment (f32 -> bf16)
            {
                const int cA = (wave * 2) * 16 + colL;
                const int cB = cA + 16;
                #pragma unroll
                for (int e = 0; e < 16; ++e) {
                    const int k = a_k_index(e, lane);
                    b0[e] = f2bf(tileF[k * TSB + cA]);
                    b1[e] = f2bf(tileF[k * TSB + cB]);
                }
            }
            a0 = __builtin_amdgcn_wmma_f32_16x16x32_bf16(false, a, false, b0, (short)0, a0, false, false);
            a1 = __builtin_amdgcn_wmma_f32_16x16x32_bf16(false, a, false, b1, (short)0, a1, false, false);
            __syncthreads();
        }
        float* wp = wpart + (size_t)(b * SPLITS + sp) * HEADS * D_DIM;
        #pragma unroll
        for (int r = 0; r < 8; ++r) {
            const int m = r + moff;
            wp[(size_t)m * D_DIM + d0c + (wave * 2 + 0) * 16 + colL] = a0[r];
            wp[(size_t)m * D_DIM + d0c + (wave * 2 + 1) * 16 + colL] = a1[r];
        }
    }
}

// ---------------------------------------------------------------------------
// K3: combine split-softmax partials -> wnorm[b,h,:] (normalized attn @ ctx)
// ---------------------------------------------------------------------------
__global__ __launch_bounds__(256) void k_combine(const float* __restrict__ wpart,
                                                 const float* __restrict__ mpart,
                                                 const float* __restrict__ lpart,
                                                 float* __restrict__ wnorm) {
    __shared__ float fac[SPLITS * HEADS];
    __shared__ float Linv[HEADS];
    const int b = blockIdx.x, tid = threadIdx.x;
    if (tid < HEADS) {
        float M = -3e38f;
        for (int s = 0; s < SPLITS; ++s)
            M = fmaxf(M, mpart[(b * SPLITS + s) * HEADS + tid]);
        float L = 0.f;
        for (int s = 0; s < SPLITS; ++s) {
            const float f = __expf(mpart[(b * SPLITS + s) * HEADS + tid] - M);
            fac[s * HEADS + tid] = f;
            L += lpart[(b * SPLITS + s) * HEADS + tid] * f;
        }
        Linv[tid] = 1.0f / L;
    }
    __syncthreads();
    for (int i = 0; i < 64; ++i) {
        const int idx = tid + i * 256;            // 0..16383
        const int h = idx >> 10, d = idx & 1023;
        float acc = 0.f;
        for (int s = 0; s < SPLITS; ++s)
            acc += wpart[((size_t)(b * SPLITS + s) * HEADS + h) * D_DIM + d] * fac[s * HEADS + h];
        wnorm[((size_t)b * HEADS + h) * D_DIM + d] = acc * Linv[h];
    }
}

// ---------------------------------------------------------------------------
// K4: attn_out[b,dout] = wnorm[b, h(dout), :] @ Wv[:, dout] + bv[dout]
// ---------------------------------------------------------------------------
__global__ __launch_bounds__(256) void k_vproj(const float* __restrict__ wnorm,
                                               const float* __restrict__ Wv,
                                               const float* __restrict__ bv,
                                               float* __restrict__ attnout) {
    const int b = blockIdx.x;
    const int dout = threadIdx.x * 4;
    const int h = dout >> 6;
    float4 acc = { bv[dout], bv[dout + 1], bv[dout + 2], bv[dout + 3] };
    const float* wn = wnorm + ((size_t)b * HEADS + h) * D_DIM;
    for (int din = 0; din < D_DIM; ++din) {
        const float wv = wn[din];
        const float4 row = *(const float4*)(Wv + (size_t)din * D_DIM + dout);
        acc.x += wv * row.x; acc.y += wv * row.y; acc.z += wv * row.z; acc.w += wv * row.w;
    }
    *(float4*)(attnout + (size_t)b * D_DIM + dout) = acc;
}

// ---------------------------------------------------------------------------
// K5: out projection + residual + LayerNorm
// ---------------------------------------------------------------------------
__global__ __launch_bounds__(256) void k_final(const float* __restrict__ attnout,
                                               const float* __restrict__ Wo,
                                               const float* __restrict__ bo,
                                               const float* __restrict__ query,
                                               const float* __restrict__ gamma,
                                               const float* __restrict__ beta,
                                               float* __restrict__ out) {
    __shared__ float rs[256];
    __shared__ float rq[256];
    const int b = blockIdx.x, tid = threadIdx.x;
    const int dout = tid * 4;
    const float* ao = attnout + (size_t)b * D_DIM;
    float4 acc = { bo[dout], bo[dout + 1], bo[dout + 2], bo[dout + 3] };
    for (int din = 0; din < D_DIM; ++din) {
        const float av = ao[din];
        const float4 row = *(const float4*)(Wo + (size_t)din * D_DIM + dout);
        acc.x += av * row.x; acc.y += av * row.y; acc.z += av * row.z; acc.w += av * row.w;
    }
    const float4 q4 = *(const float4*)(query + (size_t)b * D_DIM + dout);
    const float x0 = q4.x + acc.x, x1 = q4.y + acc.y, x2 = q4.z + acc.z, x3 = q4.w + acc.w;
    rs[tid] = x0 + x1 + x2 + x3;
    rq[tid] = x0 * x0 + x1 * x1 + x2 * x2 + x3 * x3;
    __syncthreads();
    for (int off = 128; off > 0; off >>= 1) {
        if (tid < off) { rs[tid] += rs[tid + off]; rq[tid] += rq[tid + off]; }
        __syncthreads();
    }
    const float mu = rs[0] * (1.0f / D_DIM);
    const float var = rq[0] * (1.0f / D_DIM) - mu * mu;
    const float rinv = rsqrtf(var + LN_EPS);
    float* o = out + (size_t)b * D_DIM + dout;
    o[0] = (x0 - mu) * rinv * gamma[dout + 0] + beta[dout + 0];
    o[1] = (x1 - mu) * rinv * gamma[dout + 1] + beta[dout + 1];
    o[2] = (x2 - mu) * rinv * gamma[dout + 2] + beta[dout + 2];
    o[3] = (x3 - mu) * rinv * gamma[dout + 3] + beta[dout + 3];
}

// ---------------------------------------------------------------------------
// Host launcher. Workspace layout (bytes):
//   qp      [32*1024 f32]        @ 0
//   kconst  [32*16   f32]        @ 131072
//   mpart   [32*16*16 f32]       @ 133120
//   lpart   [32*16*16 f32]       @ 165888
//   attnout [32*1024 f32]        @ 198656
//   uA      [32*32*32*16 bf16]   @ 329728
//   wnorm   [32*16*1024 f32]     @ 1378304
//   wpart   [32*16*16*1024 f32]  @ 3475456   (total ~35.3 MB)
// ---------------------------------------------------------------------------
extern "C" void kernel_launch(void* const* d_in, const int* in_sizes, int n_in,
                              void* d_out, int out_size, void* d_ws, size_t ws_size,
                              hipStream_t stream) {
    const float* query = (const float*)d_in[0];
    const float* ctx   = (const float*)d_in[1];
    const float* dist  = (const float*)d_in[2];
    const int*   mask  = (const int*)d_in[3];
    const float* Wq    = (const float*)d_in[4];
    const float* bq    = (const float*)d_in[5];
    const float* Wk    = (const float*)d_in[6];
    const float* bk    = (const float*)d_in[7];
    const float* Wv    = (const float*)d_in[8];
    const float* bv    = (const float*)d_in[9];
    const float* Wo    = (const float*)d_in[10];
    const float* bo    = (const float*)d_in[11];
    const float* gamma = (const float*)d_in[12];
    const float* beta  = (const float*)d_in[13];
    float* out = (float*)d_out;

    char* ws = (char*)d_ws;
    float*          qp      = (float*)(ws + 0);
    float*          kconst  = (float*)(ws + 131072);
    float*          mpart   = (float*)(ws + 133120);
    float*          lpart   = (float*)(ws + 165888);
    float*          attnout = (float*)(ws + 198656);
    unsigned short* uA      = (unsigned short*)(ws + 329728);
    float*          wnorm   = (float*)(ws + 1378304);
    float*          wpart   = (float*)(ws + 3475456);

    k_qproj <<<B_SZ, 256, 0, stream>>>(query, Wq, bq, qp);
    k_uA    <<<(B_SZ * 32 * 32 * 16) / 256, 256, 0, stream>>>(Wk, qp, uA);
    k_kconst<<<1, 512, 0, stream>>>(bk, qp, kconst);
    k_flash <<<dim3(SPLITS, B_SZ), 256, 0, stream>>>(ctx, dist, mask, uA, kconst,
                                                     mpart, lpart, wpart);
    k_combine<<<B_SZ, 256, 0, stream>>>(wpart, mpart, lpart, wnorm);
    k_vproj <<<B_SZ, 256, 0, stream>>>(wnorm, Wv, bv, attnout);
    k_final <<<B_SZ, 256, 0, stream>>>(attnout, Wo, bo, query, gamma, beta, out);

    (void)in_sizes; (void)n_in; (void)out_size; (void)ws_size;
}
